// HumanLikeSchemaStore_66529043415106
// MI455X (gfx1250) — matrix operand order repaired and checked by reading, not compiled, not run on hardware
//
#include <hip/hip_runtime.h>
#include <hip/hip_bf16.h>

// ---------------------------------------------------------------------------
// HumanLikeSchemaStore — fused schema retrieval for MI455X (gfx1250, wave32)
//
//   1) prep_misc   : query f32->bf16, Wr f32 -> WrT bf16 (transposed), zero cnt
//   2) prep_keys   : keys row-normalize -> kn bf16, active_count reduction
//   3) qproj       : q = query@Wr+br via v_wmma_f32_16x16x32_bf16, fused norm
//   4) sims_topk   : fused qn@kn^T GEMM (WMMA) + running top-8; key chunks
//                    double-buffered through LDS with GLOBAL_LOAD_ASYNC_TO_LDS
//                    (ASYNCcnt); fragment loads software-pipelined one K-step
//                    ahead of the WMMAs
//   5) gather_out  : validity mask + mean of gathered value rows
// ---------------------------------------------------------------------------

#define D_MODEL   1024
#define BATCH     4096
#define NSCHEMA   65536
#define TOPK      8
#define EPS       1e-8f

#define QTILE     64                    // queries per sims block
#define KCHUNK    32                    // keys per chunk
#define NITER     (NSCHEMA / KCHUNK)    // 2048

typedef __attribute__((ext_vector_type(8)))  __bf16 v8bf;
typedef __attribute__((ext_vector_type(16))) __bf16 v16bf;
typedef __attribute__((ext_vector_type(8)))  float  v8f;
typedef __attribute__((ext_vector_type(4)))  int    ai4;

__device__ __forceinline__ v8f wmma_bf16(v16bf a, v16bf b, v8f c) {
    // D(16x16,f32) = A(16x32,bf16) * B(32x16,bf16) + C
    return __builtin_amdgcn_wmma_f32_16x16x32_bf16(
        false, a, false, b, (short)0, c, false, false);
}

// ---- CDNA5 async global->LDS copy (ASYNCcnt-tracked), 16B per lane ---------
// Builtin signature (from compiler diagnostic): (v4i as1* src, v4i as3* dst,
// imm offset, imm cpol). Low 32 bits of a generic LDS pointer == LDS offset.
__device__ __forceinline__ void async_copy_b128(const void* g, void* l) {
#if __has_builtin(__builtin_amdgcn_global_load_async_to_lds_b128)
    __builtin_amdgcn_global_load_async_to_lds_b128(
        (__attribute__((address_space(1))) ai4*)(uintptr_t)g,
        (__attribute__((address_space(3))) ai4*)(unsigned)(uintptr_t)l,
        0, 0);
#else
    unsigned lds32 = (unsigned)(uintptr_t)l;
    asm volatile("global_load_async_to_lds_b128 %0, %1, off"
                 :: "v"(lds32), "v"((unsigned long long)(uintptr_t)g)
                 : "memory");
#endif
}

__device__ __forceinline__ void wait_async0() {
#if __has_builtin(__builtin_amdgcn_s_wait_asynccnt)
    __builtin_amdgcn_s_wait_asynccnt(0);
#else
    asm volatile("s_wait_asynccnt 0" ::: "memory");
#endif
}

// Stage KCHUNK key rows (KCHUNK * D_MODEL bf16 = 64 KB) into LDS, async.
__device__ __forceinline__ void stage_keys_async(const __bf16* gsrc,
                                                 __bf16* ldst, int tid) {
    const char* g = (const char*)gsrc;
    char* l = (char*)ldst;
    // 65536 B / 16 B = 4096 transfers; 16 per thread
    for (int i = tid; i < KCHUNK * D_MODEL * 2 / 16; i += 256)
        async_copy_b128(g + i * 16, l + i * 16);
}

// A fragment: 16x32 bf16 tile, row-major source (ld elements per row).
// lanes 0-15 -> M=lane, K = kc+0..15;
// lanes 16-31 -> M=lane-16, K = kc+16..31.
__device__ __forceinline__ v16bf make_afrag(const __bf16* base, int ld,
                                            int mbase, int kc, int lane) {
    int r = lane & 15, half = lane >> 4;
    const __bf16* p = base + (long)(mbase + r) * ld + kc + half * 16;
    return *(const v16bf*)p;
}

// B fragment: 32x16 bf16 tile of B = BT^T, BT row-major [N][K].
// lanes 0-15 -> N=lane, K=kc+0..15; lanes 16-31 -> N=lane-16, K=kc+16..31.
__device__ __forceinline__ v16bf make_bfrag(const __bf16* bt, int ld,
                                            int nbase, int kc, int lane) {
    int n = lane & 15, half = lane >> 4;
    const __bf16* p = bt + (long)(nbase + n) * ld + kc + half * 16;
    return *(const v16bf*)p;
}

// ---------------------------------------------------------------------------
// 1) query -> bf16 ; Wr -> transposed bf16 ; zero active_count
// ---------------------------------------------------------------------------
__global__ void prep_misc_kernel(const float* __restrict__ query,
                                 const float* __restrict__ Wr,
                                 __bf16* __restrict__ qbf,
                                 __bf16* __restrict__ wrt,
                                 int* __restrict__ active_count) {
    long i = (long)blockIdx.x * blockDim.x + threadIdx.x;
    if (i == 0) *active_count = 0;
    if (i < (long)BATCH * D_MODEL) qbf[i] = (__bf16)query[i];
    if (i < (long)D_MODEL * D_MODEL) {
        int n = (int)(i >> 10), k = (int)(i & (D_MODEL - 1));
        wrt[(long)n * D_MODEL + k] = (__bf16)Wr[(long)k * D_MODEL + n];
    }
}

// ---------------------------------------------------------------------------
// 2) keys row-normalize -> bf16 ; count active schemas
// ---------------------------------------------------------------------------
__global__ __launch_bounds__(256)
void prep_keys_kernel(const float* __restrict__ keys,
                      const unsigned char* __restrict__ active,
                      __bf16* __restrict__ kn,
                      int* __restrict__ active_count) {
    __shared__ float red[256];
    int r = blockIdx.x, tid = threadIdx.x;
    const float* row = keys + (long)r * D_MODEL;
    float ss = 0.f;
    for (int d = tid; d < D_MODEL; d += 256) { float v = row[d]; ss += v * v; }
    red[tid] = ss;
    __syncthreads();
    for (int s = 128; s > 0; s >>= 1) {
        if (tid < s) red[tid] += red[tid + s];
        __syncthreads();
    }
    float scale = 1.0f / fmaxf(sqrtf(red[0]), EPS);
    __bf16* orow = kn + (long)r * D_MODEL;
    for (int d = tid; d < D_MODEL; d += 256) orow[d] = (__bf16)(row[d] * scale);
    if (tid == 0 && active[r]) atomicAdd(active_count, 1);
}

// ---------------------------------------------------------------------------
// 3) q = query@Wr + br (WMMA), fused row-norm -> qn bf16
// ---------------------------------------------------------------------------
__global__ __launch_bounds__(256)
void qproj_kernel(const __bf16* __restrict__ qbf,
                  const __bf16* __restrict__ wrt,
                  const float* __restrict__ br,
                  __bf16* __restrict__ qn) {
    extern __shared__ char smem[];
    float* qtile = (float*)smem;                       // 16*1024 f32
    float* red   = (float*)(smem + 16 * D_MODEL * 4);  // 16*16
    float* scale = red + 256;                          // 16

    int tid = threadIdx.x;
    int wave = tid >> 5, lane = tid & 31, half = lane >> 4;
    int mbase = blockIdx.x * 16;

    for (int nt = 0; nt < 8; ++nt) {
        int nb = wave * 128 + nt * 16;
        v8f acc = {};
        // Software-pipelined: fragments for kc+32 load before the kc WMMA.
        v16bf a_nxt = make_afrag(qbf, D_MODEL, mbase, 0, lane);
        v16bf b_nxt = make_bfrag(wrt, D_MODEL, nb, 0, lane);
        #pragma unroll
        for (int kc = 0; kc < D_MODEL; kc += 32) {
            v16bf a_cur = a_nxt, b_cur = b_nxt;
            if (kc + 32 < D_MODEL) {
                a_nxt = make_afrag(qbf, D_MODEL, mbase, kc + 32, lane);
                b_nxt = make_bfrag(wrt, D_MODEL, nb, kc + 32, lane);
            }
            acc = wmma_bf16(a_cur, b_cur, acc);
        }
        int ncol = nb + (lane & 15);
        float bias = br[ncol];
        for (int j = 0; j < 8; ++j) {
            int m = half * 8 + j;
            qtile[m * D_MODEL + ncol] = acc[j] + bias;
        }
    }
    __syncthreads();

    int r = tid >> 4, seg = tid & 15;
    float ss = 0.f;
    for (int c2 = seg * 64; c2 < seg * 64 + 64; ++c2) {
        float v = qtile[r * D_MODEL + c2];
        ss += v * v;
    }
    red[r * 16 + seg] = ss;
    __syncthreads();
    if (tid < 16) {
        float s = 0.f;
        for (int j = 0; j < 16; ++j) s += red[tid * 16 + j];
        scale[tid] = 1.0f / fmaxf(sqrtf(s), EPS);
    }
    __syncthreads();
    for (int i = tid; i < 16 * D_MODEL; i += 256) {
        int m = i >> 10, c2 = i & (D_MODEL - 1);
        qn[(long)(mbase + m) * D_MODEL + c2] = (__bf16)(qtile[i] * scale[m]);
    }
}

// ---------------------------------------------------------------------------
// 4) Fused sims GEMM + top-8, async double-buffered key streaming.
//    64 queries / block (64 blocks). 8 waves: wave w -> query tile (w>>1),
//    key tile (w&1) of the 32-key chunk. LDS: qs 128KB + 2x64KB ks + 8KB sc.
// ---------------------------------------------------------------------------
#define SIMS_SMEM (QTILE * D_MODEL * 2 + 2 * KCHUNK * D_MODEL * 2 + QTILE * KCHUNK * 4)

__global__ __launch_bounds__(256)
void sims_topk_kernel(const __bf16* __restrict__ qn,
                      const __bf16* __restrict__ kn,
                      const unsigned char* __restrict__ active,
                      int* __restrict__ top_idx) {
    extern __shared__ char smem[];
    __bf16* qs  = (__bf16*)smem;                                    // 64x1024
    __bf16* ksb = (__bf16*)(smem + QTILE * D_MODEL * 2);            // 2x32x1024
    float*  sc  = (float*)(smem + QTILE * D_MODEL * 2
                                + 2 * KCHUNK * D_MODEL * 2);        // 64x32

    int tid = threadIdx.x;
    int wave = tid >> 5, lane = tid & 31, half = lane >> 4;
    int qt = wave >> 1, kt = wave & 1;
    int qbase = blockIdx.x * QTILE;
    int mb = qt * 16;

    // Stage the block's 64 query rows into LDS once.
    {
        const v8bf* s = (const v8bf*)(qn + (long)qbase * D_MODEL);
        v8bf* d = (v8bf*)qs;
        for (int i = tid; i < QTILE * D_MODEL / 8; i += 256) d[i] = s[i];
    }

    // Prologue: async-stage chunk 0 into buffer 0.
    stage_keys_async(kn, ksb, tid);
    wait_async0();

    float s8[TOPK];
    int   i8[TOPK];
    for (int j = 0; j < TOPK; ++j) { s8[j] = -1e30f; i8[j] = 0; }
    __syncthreads();

    for (int c = 0; c < NITER; ++c) {
        int kb = c * KCHUNK;
        __bf16* kcur = ksb + (c & 1) * (KCHUNK * D_MODEL);
        __bf16* knxt = ksb + ((c + 1) & 1) * (KCHUNK * D_MODEL);

        // Kick off async staging of the NEXT chunk; overlaps the WMMAs below.
        if (c + 1 < NITER)
            stage_keys_async(kn + (long)(kb + KCHUNK) * D_MODEL, knxt, tid);

        // 16x16 score tile, K = 1024 (32 WMMAs), ds-loads pipelined 1 ahead.
        v8f acc = {};
        v16bf a_nxt = make_afrag(qs, D_MODEL, mb, 0, lane);
        v16bf b_nxt = make_bfrag(kcur, D_MODEL, kt * 16, 0, lane);
        #pragma unroll
        for (int kc = 0; kc < D_MODEL; kc += 32) {
            v16bf a_cur = a_nxt, b_cur = b_nxt;
            if (kc + 32 < D_MODEL) {
                a_nxt = make_afrag(qs, D_MODEL, mb, kc + 32, lane);
                b_nxt = make_bfrag(kcur, D_MODEL, kt * 16, kc + 32, lane);
            }
            acc = wmma_bf16(a_cur, b_cur, acc);
        }
        int col = kt * 16 + (lane & 15);
        float am = active[kb + col] ? 1.0f : 0.0f;
        for (int j = 0; j < 8; ++j)
            sc[(mb + half * 8 + j) * KCHUNK + col] = acc[j] * am;
        __syncthreads();

        // Running top-8: thread t owns query t>>2, scans 8 keys (t&3).
        {
            int q = tid >> 2, h = tid & 3;
            const float* row = sc + q * KCHUNK + h * 8;
            for (int e = 0; e < 8; ++e) {
                float s = row[e];
                if (s > s8[TOPK - 1]) {
                    int idx = kb + h * 8 + e;
                    int p = TOPK - 1;
                    while (p > 0 && s8[p - 1] < s) {
                        s8[p] = s8[p - 1]; i8[p] = i8[p - 1]; --p;
                    }
                    s8[p] = s; i8[p] = idx;
                }
            }
        }

        // Next buffer must be resident before the next iteration's WMMAs.
        wait_async0();
        __syncthreads();
    }

    // Merge the 4 per-query candidate lists (reuse qs region as scratch).
    float* ms = (float*)qs;                   // 64*32 scores
    int*   mi = (int*)qs + QTILE * KCHUNK;    // 64*32 indices
    int q = tid >> 2, h = tid & 3;
    for (int j = 0; j < TOPK; ++j) {
        ms[q * KCHUNK + h * 8 + j] = s8[j];
        mi[q * KCHUNK + h * 8 + j] = i8[j];
    }
    __syncthreads();
    if (h == 0) {
        bool used[KCHUNK];
        for (int e = 0; e < KCHUNK; ++e) used[e] = false;
        for (int j = 0; j < TOPK; ++j) {
            int best = -1;
            for (int e = 0; e < KCHUNK; ++e) {
                if (used[e]) continue;
                if (best < 0) { best = e; continue; }
                float se = ms[q * KCHUNK + e], sb = ms[q * KCHUNK + best];
                if (se > sb ||
                    (se == sb && mi[q * KCHUNK + e] < mi[q * KCHUNK + best]))
                    best = e;
            }
            used[best] = true;
            top_idx[(long)(qbase + q) * TOPK + j] = mi[q * KCHUNK + best];
        }
    }
}

// ---------------------------------------------------------------------------
// 5) Gather + mean of valid value rows.
// ---------------------------------------------------------------------------
__global__ __launch_bounds__(256)
void gather_out_kernel(const float* __restrict__ values,
                       const unsigned char* __restrict__ active,
                       const int* __restrict__ top_idx,
                       const int* __restrict__ topk_p,
                       const int* __restrict__ active_count,
                       float* __restrict__ out) {
    __shared__ int vidx[TOPK];
    __shared__ int vcnt;
    int b = blockIdx.x, tid = threadIdx.x;
    if (tid == 0) {
        int tk = topk_p[0];
        int ac = active_count[0];
        int k_eff = tk < ac ? tk : ac;
        int cnt = 0;
        for (int j = 0; j < TOPK; ++j) {
            int idx = top_idx[(long)b * TOPK + j];
            if (j < k_eff && active[idx]) vidx[cnt++] = idx;
        }
        vcnt = cnt;
    }
    __syncthreads();
    int cnt = vcnt;
    float inv = cnt > 0 ? 1.0f / (float)cnt : 0.0f;
    for (int d = tid; d < D_MODEL; d += 256) {
        float s = 0.f;
        for (int j = 0; j < cnt; ++j)
            s += values[(long)vidx[j] * D_MODEL + d];
        out[(long)b * D_MODEL + d] = s * inv;
    }
}

// ---------------------------------------------------------------------------
// Host side
// ---------------------------------------------------------------------------
extern "C" void kernel_launch(void* const* d_in, const int* in_sizes, int n_in,
                              void* d_out, int out_size, void* d_ws, size_t ws_size,
                              hipStream_t stream) {
    const float* query          = (const float*)d_in[0];
    const float* keys           = (const float*)d_in[1];
    const float* values         = (const float*)d_in[2];
    const float* Wr             = (const float*)d_in[3];
    const float* br             = (const float*)d_in[4];
    const unsigned char* active = (const unsigned char*)d_in[5]; // bool array
    const int*   topk_p         = (const int*)d_in[6];
    float* out = (float*)d_out;
    (void)in_sizes; (void)n_in; (void)out_size; (void)ws_size;

    // Workspace layout (bytes)
    char* ws = (char*)d_ws;
    __bf16* qbf   = (__bf16*)(ws);                                  // 8 MB
    __bf16* wrt   = (__bf16*)(ws + 8388608);                        // 2 MB
    __bf16* qn    = (__bf16*)(ws + 10485760);                       // 8 MB
    __bf16* kn    = (__bf16*)(ws + 18874368);                       // 128 MB
    int*    tidx  = (int*)   (ws + 153092096);                      // 128 KB
    int*    acnt  = (int*)   (ws + 153223168);                      // 4 B

    prep_misc_kernel<<<(BATCH * D_MODEL + 255) / 256, 256, 0, stream>>>(
        query, Wr, qbf, wrt, acnt);

    prep_keys_kernel<<<NSCHEMA, 256, 0, stream>>>(keys, active, kn, acnt);

    size_t qproj_smem = 16 * D_MODEL * 4 + 256 * 4 + 16 * 4;
    qproj_kernel<<<BATCH / 16, 256, qproj_smem, stream>>>(qbf, wrt, br, qn);

    sims_topk_kernel<<<BATCH / QTILE, 256, SIMS_SMEM, stream>>>(qn, kn, active, tidx);

    gather_out_kernel<<<BATCH, 256, 0, stream>>>(values, active, tidx, topk_p,
                                                 acnt, out);
}
